// HistoryDeepKoopmanNoDec_59227599012596
// MI455X (gfx1250) — matrix operand order repaired and checked by reading, not compiled, not run on hardware
//
#include <hip/hip_runtime.h>
#include <hip/hip_bf16.h>
#include <stdint.h>

// ---------------------------------------------------------------------------
// Problem constants
// ---------------------------------------------------------------------------
#define STATE_DIM   32
#define EMBED_DIM   96
#define LATENT_DIM  128
#define HIST_DIM    256
#define CTRL_DIM    8
#define HW          512
#define BATCH       4096
#define HORIZON     64

#define MROWS       32      // encoder rows per block
#define RROWS       64      // rollout batch rows per block

typedef __attribute__((ext_vector_type(16))) __bf16 v16bf;
typedef __attribute__((ext_vector_type(8)))  float  v8f;

union Frag16 {            // one WMMA 16-bf16-per-lane operand = 8 VGPRs
    uint4  q[2];
    v16bf  v;
};

__device__ __forceinline__ unsigned short f2bf(float f) {
    unsigned int u = __float_as_uint(f);
    unsigned int r = u + 0x7FFFu + ((u >> 16) & 1u);   // round-to-nearest-even
    return (unsigned short)(r >> 16);
}

// ---------------------------------------------------------------------------
// f32 -> bf16 weight conversion
// ---------------------------------------------------------------------------
__global__ void convert_bf16_kernel(const float* __restrict__ src,
                                    unsigned short* __restrict__ dst, int n) {
    int i = blockIdx.x * blockDim.x + threadIdx.x;
    if (i < n) dst[i] = f2bf(src[i]);
}

// ---------------------------------------------------------------------------
// One hidden MLP layer for a 32-row tile, 2 N-tiles register-blocked per wave.
//   A: LDS bf16 (MROWS x LDA row-major).  W: global bf16 (NOUT x K row-major).
//   out = relu(A @ W^T + bias) -> LDS bf16 (stride 512).
// One A-fragment feeds two independent WMMA accumulator chains.
// ---------------------------------------------------------------------------
template<int K, int LDA, int NOUT>
__device__ __forceinline__ void layer_hidden(const unsigned short* __restrict__ ldsA,
                                             unsigned short*       __restrict__ ldsOut,
                                             const unsigned short* __restrict__ Wg,
                                             const float*          __restrict__ bias,
                                             int wave, int lane) {
    const int lrow  = lane & 15;
    const int khalf = lane >> 4;
    constexpr int numN     = NOUT / 16;                    // 32
    constexpr int numPairs = (MROWS / 16) * (numN / 2);    // 32

    for (int p = wave; p < numPairs; p += 8) {
        const int mt  = p / (numN / 2);
        const int np  = p - mt * (numN / 2);
        const int nt0 = np * 2;
        v8f acc0 = {};
        v8f acc1 = {};
        const unsigned short* arow  = ldsA + (mt * 16 + lrow) * LDA + khalf * 8;
        const unsigned short* wrow0 = Wg   + (nt0 * 16 + lrow) * K  + khalf * 8;
        const unsigned short* wrow1 = wrow0 + 16 * K;
        #pragma unroll
        for (int k0 = 0; k0 < K; k0 += 32) {
            Frag16 a, b0, b1;
            a.q[0]  = *(const uint4*)(arow + k0);          // K = k0+khalf*8 .. +7
            a.q[1]  = *(const uint4*)(arow + k0 + 16);     // K = k0+16+khalf*8 .. +7
            b0.q[0] = *(const uint4*)(wrow0 + k0);         // K = k0+khalf*8 .. +7
            b0.q[1] = *(const uint4*)(wrow0 + k0 + 16);    // K = k0+16+khalf*8 .. +7
            b1.q[0] = *(const uint4*)(wrow1 + k0);
            b1.q[1] = *(const uint4*)(wrow1 + k0 + 16);
            acc0 = __builtin_amdgcn_wmma_f32_16x16x32_bf16(false, a.v, false, b0.v,
                                                           (short)0, acc0, false, false);
            acc1 = __builtin_amdgcn_wmma_f32_16x16x32_bf16(false, a.v, false, b1.v,
                                                           (short)0, acc1, false, false);
        }
        const int   n0  = nt0 * 16 + lrow;
        const float bv0 = bias[n0];
        const float bv1 = bias[n0 + 16];
        #pragma unroll
        for (int v = 0; v < 8; ++v) {
            const int m = mt * 16 + khalf * 8 + v;
            float f0 = acc0[v] + bv0;
            float f1 = acc1[v] + bv1;
            f0 = f0 > 0.0f ? f0 : 0.0f;
            f1 = f1 > 0.0f ? f1 : 0.0f;
            ldsOut[m * 512 + n0]      = f2bf(f0);
            ldsOut[m * 512 + n0 + 16] = f2bf(f1);
        }
    }
}

// Final layer: K=512, NOUT=96, no relu, f32 out at columns 32..127 of z row.
// 6 waves, each owning one N-tile; the B-fragment is shared across both M-tiles.
__device__ __forceinline__ void layer_final(const unsigned short* __restrict__ ldsA,
                                            float*                __restrict__ outz,
                                            const unsigned short* __restrict__ Wg,
                                            const float*          __restrict__ bias,
                                            int wave, int lane) {
    const int lrow  = lane & 15;
    const int khalf = lane >> 4;
    constexpr int numN = EMBED_DIM / 16;                   // 6

    if (wave < numN) {
        const int nt = wave;
        v8f acc0 = {};                                     // M-tile 0
        v8f acc1 = {};                                     // M-tile 1
        const unsigned short* arow0 = ldsA + lrow * 512 + khalf * 8;
        const unsigned short* arow1 = arow0 + 16 * 512;
        const unsigned short* wrow  = Wg + (nt * 16 + lrow) * 512 + khalf * 8;
        #pragma unroll
        for (int k0 = 0; k0 < 512; k0 += 32) {
            Frag16 a0, a1, b;
            b.q[0]  = *(const uint4*)(wrow + k0);
            b.q[1]  = *(const uint4*)(wrow + k0 + 16);
            a0.q[0] = *(const uint4*)(arow0 + k0);
            a0.q[1] = *(const uint4*)(arow0 + k0 + 16);
            a1.q[0] = *(const uint4*)(arow1 + k0);
            a1.q[1] = *(const uint4*)(arow1 + k0 + 16);
            acc0 = __builtin_amdgcn_wmma_f32_16x16x32_bf16(false, a0.v, false, b.v,
                                                           (short)0, acc0, false, false);
            acc1 = __builtin_amdgcn_wmma_f32_16x16x32_bf16(false, a1.v, false, b.v,
                                                           (short)0, acc1, false, false);
        }
        const int   nIdx = nt * 16 + lrow;
        const float bv   = bias[nIdx];
        #pragma unroll
        for (int v = 0; v < 8; ++v) {
            const int m0 = khalf * 8 + v;
            outz[(size_t)m0 * LATENT_DIM + STATE_DIM + nIdx]        = acc0[v] + bv;
            outz[(size_t)(m0 + 16) * LATENT_DIM + STATE_DIM + nIdx] = acc1[v] + bv;
        }
    }
}

// ---------------------------------------------------------------------------
// Fused 4-layer encoder + lift.  Blocks [0,128): history_k -> zk workspace.
// Blocks [128, 128+8192): history_next rows -> z_target region of d_out.
// ---------------------------------------------------------------------------
__global__ __launch_bounds__(256) void encoder_kernel(
        const float* __restrict__ history_k,
        const float* __restrict__ hist_next,
        const unsigned short* __restrict__ Wb0, const float* __restrict__ b0,
        const unsigned short* __restrict__ Wb1, const float* __restrict__ b1,
        const unsigned short* __restrict__ Wb2, const float* __restrict__ b2,
        const unsigned short* __restrict__ Wb3, const float* __restrict__ b3,
        float* __restrict__ zk, float* __restrict__ ztar) {
    __shared__ unsigned short bufA[MROWS * 512];
    __shared__ unsigned short bufB[MROWS * 512];

    const int tid  = threadIdx.x;
    const int wave = tid >> 5;
    const int lane = tid & 31;

    const float* in;
    float*       outz;
    constexpr int NB_ZK = BATCH / MROWS;                 // 128
    if ((int)blockIdx.x < NB_ZK) {
        const size_t r0 = (size_t)blockIdx.x * MROWS;
        in   = history_k + r0 * HIST_DIM;
        outz = zk        + r0 * LATENT_DIM;
    } else {
        const size_t r0 = (size_t)(blockIdx.x - NB_ZK) * MROWS;
        in   = hist_next + r0 * HIST_DIM;
        outz = ztar      + r0 * LATENT_DIM;
    }

    // Stage input (32 x 256 f32) -> bf16 LDS; copy "cur" (last 32 cols) -> z[0:32].
    const float4* in4 = (const float4*)in;
    #pragma unroll
    for (int j = 0; j < (MROWS * HIST_DIM / 4) / 256; ++j) {   // 8 iters
        const int f4  = tid + j * 256;
        const int row = f4 >> 6;
        const int c4  = f4 & 63;
        const float4 val = in4[f4];
        const int base = row * HIST_DIM + c4 * 4;
        bufA[base + 0] = f2bf(val.x);
        bufA[base + 1] = f2bf(val.y);
        bufA[base + 2] = f2bf(val.z);
        bufA[base + 3] = f2bf(val.w);
        if (c4 >= 56) {                                  // cols 224..255 -> cur
            float* o = outz + (size_t)row * LATENT_DIM + (c4 * 4 - 224);
            o[0] = val.x; o[1] = val.y; o[2] = val.z; o[3] = val.w;
        }
    }
    __syncthreads();
    layer_hidden<HIST_DIM, HIST_DIM, HW>(bufA, bufB, Wb0, b0, wave, lane);
    __syncthreads();
    layer_hidden<HW, HW, HW>(bufB, bufA, Wb1, b1, wave, lane);
    __syncthreads();
    layer_hidden<HW, HW, HW>(bufA, bufB, Wb2, b2, wave, lane);
    __syncthreads();
    layer_final(bufB, outz, Wb3, b3, wave, lane);
}

// ---------------------------------------------------------------------------
// Fused 64-step Koopman rollout: z <- z @ A^T + bu_t, all steps in-kernel.
// Block = 256 threads, 64 batch rows; wave w owns latent columns [16w,16w+16).
// A^T fragments and B_W row live in registers for the whole rollout.
// ---------------------------------------------------------------------------
__global__ __launch_bounds__(256) void rollout_kernel(
        const float* __restrict__ zk,
        const float* __restrict__ u_seq,
        const float* __restrict__ B_W,
        const unsigned short* __restrict__ Ab,   // bf16 A_W, (128 x 128) row-major
        float* __restrict__ z_pred,
        float* __restrict__ x_pred) {
    __shared__ unsigned short zbuf[2][RROWS * LATENT_DIM];   // 32 KB
    __shared__ float          ustage[2][RROWS * CTRL_DIM];   // 4 KB

    const int tid   = threadIdx.x;
    const int wave  = tid >> 5;
    const int lane  = tid & 31;
    const int lrow  = lane & 15;
    const int khalf = lane >> 4;
    const int rowBase = blockIdx.x * RROWS;
    const int n = wave * 16 + lrow;                 // this lane's latent column

    // init z state from zk (f32 -> bf16)
    #pragma unroll
    for (int j = 0; j < (RROWS * LATENT_DIM) / 256; ++j) {   // 32 iters
        const int f = tid + j * 256;
        zbuf[0][f] = f2bf(zk[(size_t)rowBase * LATENT_DIM + f]);
    }
    // stage u for t=0
    {
        const int idx = tid * 2;
        const int m = idx >> 3, c = idx & 7;
        const float* up = u_seq + ((size_t)(rowBase + m) * HORIZON + 0) * CTRL_DIM + c;
        ustage[0][idx]     = up[0];
        ustage[0][idx + 1] = up[1];
    }

    // Preload B-fragments of A^T (B[k][n] = A_W[n][k]) and this lane's B_W row.
    Frag16 bfrag[4];
    const unsigned short* wrow = Ab + n * LATENT_DIM + khalf * 8;
    #pragma unroll
    for (int ks = 0; ks < 4; ++ks) {
        bfrag[ks].q[0] = *(const uint4*)(wrow + ks * 32);
        bfrag[ks].q[1] = *(const uint4*)(wrow + ks * 32 + 16);
    }
    float bw[CTRL_DIM];
    #pragma unroll
    for (int c = 0; c < CTRL_DIM; ++c) bw[c] = B_W[n * CTRL_DIM + c];

    __syncthreads();

    for (int t = 0; t < HORIZON; ++t) {
        const int cur = t & 1, nxt = cur ^ 1;
        if (t + 1 < HORIZON) {                       // prefetch u(t+1)
            const int idx = tid * 2;
            const int m = idx >> 3, c = idx & 7;
            const float* up = u_seq + ((size_t)(rowBase + m) * HORIZON + (t + 1)) * CTRL_DIM + c;
            ustage[nxt][idx]     = up[0];
            ustage[nxt][idx + 1] = up[1];
        }
        #pragma unroll
        for (int mt = 0; mt < 4; ++mt) {
            // accumulator seeded with bu[t, b, n] = dot(u[b,t,:], B_W[n,:])
            v8f acc;
            #pragma unroll
            for (int v = 0; v < 8; ++v) {
                const int m = mt * 16 + khalf * 8 + v;
                float s = 0.0f;
                #pragma unroll
                for (int c = 0; c < CTRL_DIM; ++c)
                    s += ustage[cur][m * CTRL_DIM + c] * bw[c];
                acc[v] = s;
            }
            const unsigned short* arow =
                &zbuf[cur][(mt * 16 + lrow) * LATENT_DIM + khalf * 8];
            #pragma unroll
            for (int ks = 0; ks < 4; ++ks) {
                Frag16 a;
                a.q[0] = *(const uint4*)(arow + ks * 32);
                a.q[1] = *(const uint4*)(arow + ks * 32 + 16);
                acc = __builtin_amdgcn_wmma_f32_16x16x32_bf16(false, a.v, false, bfrag[ks].v,
                                                              (short)0, acc, false, false);
            }
            #pragma unroll
            for (int v = 0; v < 8; ++v) {
                const int m = mt * 16 + khalf * 8 + v;
                const int b = rowBase + m;
                const float f = acc[v];
                z_pred[((size_t)b * HORIZON + t) * LATENT_DIM + n] = f;
                if (n < STATE_DIM)
                    x_pred[((size_t)b * HORIZON + t) * STATE_DIM + n] = f;
                zbuf[nxt][m * LATENT_DIM + n] = f2bf(f);
            }
        }
        __syncthreads();
    }
}

// ---------------------------------------------------------------------------
// Host launcher
// ---------------------------------------------------------------------------
extern "C" void kernel_launch(void* const* d_in, const int* in_sizes, int n_in,
                              void* d_out, int out_size, void* d_ws, size_t ws_size,
                              hipStream_t stream) {
    const float* history_k   = (const float*)d_in[0];
    const float* u_seq       = (const float*)d_in[1];
    const float* hist_next   = (const float*)d_in[2];
    const float* enc_W0      = (const float*)d_in[3];
    const float* enc_b0      = (const float*)d_in[4];
    const float* enc_W1      = (const float*)d_in[5];
    const float* enc_b1      = (const float*)d_in[6];
    const float* enc_W2      = (const float*)d_in[7];
    const float* enc_b2      = (const float*)d_in[8];
    const float* enc_W3      = (const float*)d_in[9];
    const float* enc_b3      = (const float*)d_in[10];
    const float* A_W         = (const float*)d_in[11];
    const float* B_W         = (const float*)d_in[12];

    float* z_pred   = (float*)d_out;                                   // 4096*64*128
    float* x_pred   = z_pred + (size_t)BATCH * HORIZON * LATENT_DIM;   // 4096*64*32
    float* z_target = x_pred + (size_t)BATCH * HORIZON * STATE_DIM;    // 4096*64*128

    // Workspace layout (bytes)
    char* ws = (char*)d_ws;
    unsigned short* Wb0 = (unsigned short*)(ws);                  // 512*256 bf16
    unsigned short* Wb1 = (unsigned short*)(ws + 262144);         // 512*512 bf16
    unsigned short* Wb2 = (unsigned short*)(ws + 262144 + 524288);
    unsigned short* Wb3 = (unsigned short*)(ws + 262144 + 524288 + 524288); // 96*512
    unsigned short* Ab  = (unsigned short*)(ws + 262144 + 524288 + 524288 + 98304); // 128*128
    float*          zk  = (float*)        (ws + 262144 + 524288 + 524288 + 98304 + 32768);

    // Weight conversion to bf16
    auto cvt = [&](const float* src, unsigned short* dst, int n) {
        convert_bf16_kernel<<<(n + 255) / 256, 256, 0, stream>>>(src, dst, n);
    };
    cvt(enc_W0, Wb0, HW * HIST_DIM);
    cvt(enc_W1, Wb1, HW * HW);
    cvt(enc_W2, Wb2, HW * HW);
    cvt(enc_W3, Wb3, EMBED_DIM * HW);
    cvt(A_W,    Ab,  LATENT_DIM * LATENT_DIM);

    // Encoder: 128 blocks for z_k + 8192 blocks for z_target rows
    const int nblk = BATCH / MROWS + (BATCH * HORIZON) / MROWS;   // 128 + 8192
    encoder_kernel<<<nblk, 256, 0, stream>>>(history_k, hist_next,
                                             Wb0, enc_b0, Wb1, enc_b1,
                                             Wb2, enc_b2, Wb3, enc_b3,
                                             zk, z_target);

    // Fused 64-step rollout
    rollout_kernel<<<BATCH / RROWS, 256, 0, stream>>>(zk, u_seq, B_W, Ab,
                                                      z_pred, x_pred);
}
